// Net_44255343018662
// MI455X (gfx1250) — matrix operand order repaired and checked by reading, not compiled, not run on hardware
//
#include <hip/hip_runtime.h>
#include <hip/hip_bf16.h>
#include <math.h>

#define IN_DIM 512
#define HID 16
#define NCLS 7

typedef __attribute__((ext_vector_type(16))) __bf16 v16bf;
typedef __attribute__((ext_vector_type(8)))  float  v8f;

__device__ __forceinline__ void split_bf16(float f, __bf16& hi, __bf16& lo) {
    hi = (__bf16)f;
    lo = (__bf16)(f - (float)hi);
}

// ---------------------------------------------------------------------------
// Degree / normalization
// ---------------------------------------------------------------------------
__global__ void deg_kernel(const int* __restrict__ dst, unsigned* __restrict__ deg, int E) {
    int e = blockIdx.x * blockDim.x + threadIdx.x;
    if (e < E) atomicAdd(&deg[dst[e]], 1u);
}

__global__ void dinv_kernel(const unsigned* __restrict__ deg, float* __restrict__ dinv, int n) {
    int i = blockIdx.x * blockDim.x + threadIdx.x;
    if (i < n) dinv[i] = rsqrtf((float)deg[i] + 1.0f);   // +1 = self loop
}

// ---------------------------------------------------------------------------
// GEMM1: h = x @ W1   (16x16 tile per wave, K=512, split-bf16 WMMA, fp32 acc)
// A layout (16-bit 16x32): lane = M (mod 16), half-select hs = lane>>4,
//   elems 0..7 -> K = k0 + 8*hs + 0..7 ; elems 8..15 -> K = k0 + 16 + 8*hs + 0..7
// B layout (dense 32x16): lane = N (mod 16), elem e -> same K as A elem e
// C/D layout: lane n = lane&15, VGPR j -> M = j + 8*hs
// ---------------------------------------------------------------------------
__global__ void gemm1_wmma(const float* __restrict__ x, const float* __restrict__ W1,
                           float* __restrict__ hout, int nnodes) {
    const int lane = threadIdx.x & 31;
    const int wid  = threadIdx.x >> 5;
    const int tile = blockIdx.x * (blockDim.x >> 5) + wid;
    const int ntiles = (nnodes + 15) >> 4;
    if (tile >= ntiles) return;

    const int m  = lane & 15;
    const int hs = lane >> 4;
    int row = tile * 16 + m;
    if (row >= nnodes) row = nnodes - 1;
    const float* xrow = x + (size_t)row * IN_DIM;

    v8f acc = {};
    for (int k0 = 0; k0 < IN_DIM; k0 += 32) {
        float af[16], bfv[16];
        {
            const float4* p0 = (const float4*)(xrow + k0 + 8 * hs);
            float4 q0 = p0[0], q1 = p0[1];
            const float4* p1 = (const float4*)(xrow + k0 + 16 + 8 * hs);
            float4 q2 = p1[0], q3 = p1[1];
            af[0]=q0.x; af[1]=q0.y; af[2]=q0.z;  af[3]=q0.w;
            af[4]=q1.x; af[5]=q1.y; af[6]=q1.z;  af[7]=q1.w;
            af[8]=q2.x; af[9]=q2.y; af[10]=q2.z; af[11]=q2.w;
            af[12]=q3.x; af[13]=q3.y; af[14]=q3.z; af[15]=q3.w;
        }
        {
            #pragma unroll
            for (int e = 0; e < 16; ++e) {
                int kk = (e < 8) ? (8 * hs + e) : (16 + 8 * hs + (e - 8));
                bfv[e] = W1[(size_t)(k0 + kk) * HID + m];
            }
        }
        v16bf ah, al, bh, bl;
        #pragma unroll
        for (int e = 0; e < 16; ++e) {
            __bf16 hi, lo;
            split_bf16(af[e], hi, lo);  ah[e] = hi; al[e] = lo;
            split_bf16(bfv[e], hi, lo); bh[e] = hi; bl[e] = lo;
        }
        acc = __builtin_amdgcn_wmma_f32_16x16x32_bf16(false, ah, false, bh, (short)0, acc, false, false);
        acc = __builtin_amdgcn_wmma_f32_16x16x32_bf16(false, ah, false, bl, (short)0, acc, false, false);
        acc = __builtin_amdgcn_wmma_f32_16x16x32_bf16(false, al, false, bh, (short)0, acc, false, false);
    }

    float* outp = hout + (size_t)(tile * 16 + 8 * hs) * HID + m;
    if (tile * 16 + 16 <= nnodes) {           // wave-uniform fast path
        #pragma unroll
        for (int j = 0; j < 8; ++j) outp[(size_t)j * HID] = acc[j];
    } else {                                  // tail tile
        #pragma unroll
        for (int j = 0; j < 8; ++j) {
            int rm = tile * 16 + j + 8 * hs;
            if (rm < nnodes) hout[(size_t)rm * HID + m] = acc[j];
        }
    }
}

// ---------------------------------------------------------------------------
// Layer-1 scatter: agg1[dst] += h[src] * dinv[src]*dinv[dst]  (16 lanes/edge)
// ---------------------------------------------------------------------------
__global__ void scatter1(const int* __restrict__ src, const int* __restrict__ dst,
                         const float* __restrict__ dinv, const float* __restrict__ hmat,
                         float* __restrict__ agg, int E) {
    int t = blockIdx.x * blockDim.x + threadIdx.x;
    int e = t >> 4;
    if (e >= E) return;
    int c = t & 15;
    int s = src[e], d = dst[e];
    float w = dinv[s] * dinv[d];
    atomicAdd(&agg[(size_t)d * HID + c], hmat[(size_t)s * HID + c] * w);
}

// h1 = relu(agg1 + h*dinv^2 + b1)   (written in place over h)
__global__ void finalize1(float* __restrict__ hmat, const float* __restrict__ agg,
                          const float* __restrict__ dinv, const float* __restrict__ b1,
                          int total) {
    int t = blockIdx.x * blockDim.x + threadIdx.x;
    if (t >= total) return;
    int i = t >> 4, c = t & 15;
    float di = dinv[i];
    float v = agg[t] + hmat[t] * di * di + b1[c];
    hmat[t] = v > 0.f ? v : 0.f;
}

// ---------------------------------------------------------------------------
// GEMM2: z = h1 @ W2  (K=16 zero-padded to 32, N=7 zero-padded to 16,
// z stored with stride 8, column 7 = 0). Same WMMA layouts as gemm1.
// ---------------------------------------------------------------------------
__global__ void gemm2_wmma(const float* __restrict__ h1, const float* __restrict__ W2,
                           float* __restrict__ z, int nnodes) {
    const int lane = threadIdx.x & 31;
    const int wid  = threadIdx.x >> 5;
    const int tile = blockIdx.x * (blockDim.x >> 5) + wid;
    const int ntiles = (nnodes + 15) >> 4;
    if (tile >= ntiles) return;

    const int m  = lane & 15;
    const int hs = lane >> 4;
    int row = tile * 16 + m;
    if (row >= nnodes) row = nnodes - 1;
    const float* hrow = h1 + (size_t)row * HID;

    float af[16], bfv[16];
    {
        const float4* p0 = (const float4*)(hrow + 8 * hs);   // K = 8*hs .. 8*hs+7
        float4 q0 = p0[0], q1 = p0[1];
        af[0]=q0.x; af[1]=q0.y; af[2]=q0.z; af[3]=q0.w;
        af[4]=q1.x; af[5]=q1.y; af[6]=q1.z; af[7]=q1.w;
        #pragma unroll
        for (int e = 8; e < 16; ++e) af[e] = 0.f;            // K >= 16 pad
    }
    #pragma unroll
    for (int e = 0; e < 16; ++e) {
        bfv[e] = (e < 8 && m < NCLS) ? W2[(8 * hs + e) * NCLS + m] : 0.f;
    }
    v16bf ah, al, bh, bl;
    #pragma unroll
    for (int e = 0; e < 16; ++e) {
        __bf16 hi, lo;
        split_bf16(af[e], hi, lo);  ah[e] = hi; al[e] = lo;
        split_bf16(bfv[e], hi, lo); bh[e] = hi; bl[e] = lo;
    }
    v8f acc = {};
    acc = __builtin_amdgcn_wmma_f32_16x16x32_bf16(false, ah, false, bh, (short)0, acc, false, false);
    acc = __builtin_amdgcn_wmma_f32_16x16x32_bf16(false, ah, false, bl, (short)0, acc, false, false);
    acc = __builtin_amdgcn_wmma_f32_16x16x32_bf16(false, al, false, bh, (short)0, acc, false, false);

    const bool full = (tile * 16 + 16 <= nnodes);            // wave-uniform
    if (full && m < 8) {
        float* zp = z + (size_t)(tile * 16 + 8 * hs) * 8 + m;
        #pragma unroll
        for (int j = 0; j < 8; ++j) zp[(size_t)j * 8] = (m < NCLS) ? acc[j] : 0.f;
    } else if (!full) {
        #pragma unroll
        for (int j = 0; j < 8; ++j) {
            int rm = tile * 16 + j + 8 * hs;
            if (rm < nnodes && m < 8) z[(size_t)rm * 8 + m] = (m < NCLS) ? acc[j] : 0.f;
        }
    }
}

// Layer-2 scatter: agg2[dst] += z[src] * norm  (8 lanes/edge, lane 7 idle)
__global__ void scatter2(const int* __restrict__ src, const int* __restrict__ dst,
                         const float* __restrict__ dinv, const float* __restrict__ z,
                         float* __restrict__ agg2, int E) {
    int t = blockIdx.x * blockDim.x + threadIdx.x;
    int e = t >> 3;
    if (e >= E) return;
    int c = t & 7;
    if (c < NCLS) {
        int s = src[e], d = dst[e];
        float w = dinv[s] * dinv[d];
        atomicAdd(&agg2[(size_t)d * 8 + c], z[(size_t)s * 8 + c] * w);
    }
}

// out = log_softmax(agg2 + z*dinv^2 + b2)
__global__ void final_kernel(const float* __restrict__ agg2, const float* __restrict__ z,
                             const float* __restrict__ dinv, const float* __restrict__ b2,
                             float* __restrict__ out, int n) {
    int i = blockIdx.x * blockDim.x + threadIdx.x;
    if (i >= n) return;
    float di = dinv[i], dii = di * di;
    float v[NCLS], mx = -1e30f;
    #pragma unroll
    for (int c = 0; c < NCLS; ++c) {
        v[c] = agg2[(size_t)i * 8 + c] + z[(size_t)i * 8 + c] * dii + b2[c];
        mx = fmaxf(mx, v[c]);
    }
    float s = 0.f;
    #pragma unroll
    for (int c = 0; c < NCLS; ++c) s += expf(v[c] - mx);
    float ls = logf(s);
    #pragma unroll
    for (int c = 0; c < NCLS; ++c) out[(size_t)i * NCLS + c] = v[c] - mx - ls;
}

// ---------------------------------------------------------------------------
extern "C" void kernel_launch(void* const* d_in, const int* in_sizes, int n_in,
                              void* d_out, int out_size, void* d_ws, size_t ws_size,
                              hipStream_t stream) {
    const float* x  = (const float*)d_in[0];
    const int*   ei = (const int*)d_in[1];
    const float* W1 = (const float*)d_in[2];
    const float* b1 = (const float*)d_in[3];
    const float* W2 = (const float*)d_in[4];
    const float* b2 = (const float*)d_in[5];
    float* out = (float*)d_out;

    const int N = in_sizes[0] / IN_DIM;
    const int E = in_sizes[1] / 2;
    const int* src = ei;
    const int* dst = ei + E;

    char* ws = (char*)d_ws;
    size_t off = 0;
    auto alloc = [&](size_t bytes) -> char* {
        char* p = ws + off;
        off = (off + bytes + 255) & ~(size_t)255;
        return p;
    };
    unsigned* deg  = (unsigned*)alloc((size_t)N * 4);
    float*    dinv = (float*)   alloc((size_t)N * 4);
    float*    hbuf = (float*)   alloc((size_t)N * HID * 4);
    float*    agg1 = (float*)   alloc((size_t)N * HID * 4);
    float*    zbuf = (float*)   alloc((size_t)N * 8 * 4);
    float*    agg2 = (float*)   alloc((size_t)N * 8 * 4);

    hipMemsetAsync(deg,  0, (size_t)N * 4,       stream);
    hipMemsetAsync(agg1, 0, (size_t)N * HID * 4, stream);
    hipMemsetAsync(agg2, 0, (size_t)N * 8 * 4,   stream);

    const int thr = 256;
    deg_kernel <<<(E + thr - 1) / thr, thr, 0, stream>>>(dst, deg, E);
    dinv_kernel<<<(N + thr - 1) / thr, thr, 0, stream>>>(deg, dinv, N);

    const int ntiles = (N + 15) / 16;
    gemm1_wmma<<<(ntiles + 3) / 4, 128, 0, stream>>>(x, W1, hbuf, N);

    long long t1 = (long long)E * 16;
    scatter1<<<(unsigned)((t1 + thr - 1) / thr), thr, 0, stream>>>(src, dst, dinv, hbuf, agg1, E);

    int t2 = N * HID;
    finalize1<<<(t2 + thr - 1) / thr, thr, 0, stream>>>(hbuf, agg1, dinv, b1, t2);

    gemm2_wmma<<<(ntiles + 3) / 4, 128, 0, stream>>>(hbuf, W2, zbuf, N);

    long long t3 = (long long)E * 8;
    scatter2<<<(unsigned)((t3 + thr - 1) / thr), thr, 0, stream>>>(src, dst, dinv, zbuf, agg2, E);

    final_kernel<<<(N + thr - 1) / thr, thr, 0, stream>>>(agg2, zbuf, dinv, b2, out, N);
}
